// RGCN_54417235641086
// MI455X (gfx1250) — matrix operand order repaired and checked by reading, not compiled, not run on hardware
//
#include <hip/hip_runtime.h>
#include <math.h>

#define N_USER 40000
#define N_ITEM 40000
#define N_EDGE 250000
#define IN_FEATS 128
#define HID 256
#define OUTF 128
#define BN_EPS 1e-5f

typedef __attribute__((ext_vector_type(16))) _Float16 v16h;
typedef __attribute__((ext_vector_type(8)))  _Float16 v8h;
typedef __attribute__((ext_vector_type(8)))  float    v8f;

// ---------------------------------------------------------------------------
// float atomic-max via signed/unsigned integer ordering trick
// ---------------------------------------------------------------------------
__device__ __forceinline__ void atomic_max_float(float* addr, float val) {
  if (val >= 0.0f)
    atomicMax((int*)addr, __float_as_int(val));
  else
    atomicMin((unsigned int*)addr, (unsigned int)__float_as_int(val));
}

// ---------------------------------------------------------------------------
// A fragment loader: 16-bit A-matrix 16x32 (MxK) WMMA layout.
//   lanes 0-15 : M=lane,    halves 0-7 -> K=ka..ka+7, halves 8-15 -> K=ka+16..ka+23
//   lanes 16-31: M=lane-16, ka is already offset by +8 for the hi half-wave
// Loads 4 coalesced float4 (K contiguous) and converts f32 -> f16.
// ---------------------------------------------------------------------------
__device__ __forceinline__ v16h load_a_frag(const float* __restrict__ A,
                                            long arow, int K, int ka) {
  const float4* ap = (const float4*)(A + arow * K + ka);
  float4 a0 = ap[0];   // K = ka+0..3
  float4 a1 = ap[1];   // K = ka+4..7
  float4 a2 = ap[4];   // K = ka+16..19
  float4 a3 = ap[5];   // K = ka+20..23
  v16h af;
  af[0]  = (_Float16)a0.x; af[1]  = (_Float16)a0.y;
  af[2]  = (_Float16)a0.z; af[3]  = (_Float16)a0.w;
  af[4]  = (_Float16)a1.x; af[5]  = (_Float16)a1.y;
  af[6]  = (_Float16)a1.z; af[7]  = (_Float16)a1.w;
  af[8]  = (_Float16)a2.x; af[9]  = (_Float16)a2.y;
  af[10] = (_Float16)a2.z; af[11] = (_Float16)a2.w;
  af[12] = (_Float16)a3.x; af[13] = (_Float16)a3.y;
  af[14] = (_Float16)a3.z; af[15] = (_Float16)a3.w;
  return af;
}

// ---------------------------------------------------------------------------
// WMMA GEMM: C[N,M] = A[N,K] * B[K,M]; fp32 in/out, f16 WMMA, f32 accumulate.
// Block = 128 threads = 4 waves. Grid = (N/32, M/64).
// Each wave computes a 32x16 tile (two 16x16 WMMAs sharing one B fragment).
// B tile (32 x 64 fp32) is staged cooperatively into LDS per K-step:
//   - coalesced float4 global reads, converted to f16, stored TRANSPOSED
//     ( Bs[col][k] ) so each lane's B fragment is two contiguous ds_load_b128.
// B fragment layout (32x16 KxN): lanes 0-15 N=lane, K=0..7,16..23; lanes 16-31 K=8..15,24..31.
// C/D layout: VGPR r -> row r (lanes 0-15) / r+8 (lanes 16-31), col = lane&15.
// ---------------------------------------------------------------------------
#define BS_STRIDE 40   // halves per LDS column (32 + 8 pad), 80B: keeps 16B align

__global__ __launch_bounds__(128) void gemm_wmma_f16(
    const float* __restrict__ A, const float* __restrict__ B,
    float* __restrict__ C, int N, int K, int M) {
  __shared__ __align__(16) _Float16 Bs[64 * BS_STRIDE];

  const int tid  = threadIdx.x;
  const int lane = tid & 31;
  const int wave = tid >> 5;
  const int l15  = lane & 15;
  const int hi   = lane >> 4;
  const int row0    = blockIdx.x * 32;
  const int colBase = blockIdx.y * 64;
  const int col0    = colBase + wave * 16;

  const long arow0 = (long)(row0 + l15);
  const long arow1 = arow0 + 16;

  v8f acc0 = {}, acc1 = {};

  for (int k0 = 0; k0 < K; k0 += 32) {
    // ---- cooperative stage of B[k0..k0+31, colBase..colBase+63] -> LDS (f16,T)
    int linear = tid;
#pragma unroll
    for (int i = 0; i < 4; ++i, linear += 128) {   // 512 float4 = 32x64 floats
      const int r  = linear >> 4;                  // k offset   0..31
      const int c4 = (linear & 15) << 2;           // col offset 0..60
      const float4 v = *(const float4*)(B + (long)(k0 + r) * M + colBase + c4);
      Bs[(c4 + 0) * BS_STRIDE + r] = (_Float16)v.x;
      Bs[(c4 + 1) * BS_STRIDE + r] = (_Float16)v.y;
      Bs[(c4 + 2) * BS_STRIDE + r] = (_Float16)v.z;
      Bs[(c4 + 3) * BS_STRIDE + r] = (_Float16)v.w;
    }
    __syncthreads();

    // ---- A fragments (coalesced float4 global loads, f32->f16)
    const int ka = k0 + hi * 8;
    v16h af0 = load_a_frag(A, arow0, K, ka);
    v16h af1 = load_a_frag(A, arow1, K, ka);

    // prefetch next K-step of A (global_prefetch_b8; speculative, OOB dropped)
    if (k0 + 32 < K) {
      __builtin_prefetch(A + arow0 * K + ka + 32, 0, 3);
      __builtin_prefetch(A + arow1 * K + ka + 32, 0, 3);
    }

    // ---- B fragment: two 8-half runs per lane (K=8h..8h+7, 16+8h..) from transposed LDS tile
    const _Float16* bp = &Bs[(wave * 16 + l15) * BS_STRIDE + hi * 8];
    v8h b0 = *(const v8h*)bp;
    v8h b1 = *(const v8h*)(bp + 16);
    v16h bf;
#pragma unroll
    for (int h = 0; h < 8; ++h) { bf[h] = b0[h]; bf[h + 8] = b1[h]; }

    // ---- two independent WMMAs sharing the B fragment
    acc0 = __builtin_amdgcn_wmma_f32_16x16x32_f16(
        false, af0, false, bf, (short)0, acc0, false, false);
    acc1 = __builtin_amdgcn_wmma_f32_16x16x32_f16(
        false, af1, false, bf, (short)0, acc1, false, false);
    __syncthreads();
  }

#pragma unroll
  for (int r = 0; r < 8; ++r) {
    C[(long)(row0 + r + hi * 8) * M + col0 + l15]      = acc0[r];
    C[(long)(row0 + 16 + r + hi * 8) * M + col0 + l15] = acc1[r];
  }
}

// ---------------------------------------------------------------------------
// el[n,h] = sum_d pfs[n, h*D+d] * al[h*D+d]
// ---------------------------------------------------------------------------
__global__ void el_reduce(const float* __restrict__ pfs, const float* __restrict__ al,
                          float* __restrict__ el, int N, int H, int D) {
  int t = blockIdx.x * blockDim.x + threadIdx.x;
  if (t >= N * H) return;
  int n = t / H, h = t % H;
  const float* p = pfs + (long)n * H * D + h * D;
  const float* a = al + h * D;
  float s = 0.f;
  for (int d = 0; d < D; ++d) s += p[d] * a[d];
  el[t] = s;
}

// ---------------------------------------------------------------------------
// wf[k,h] = sum_d W[k, h*D+d] * ar[h*D+d]     (fold attention vec into W)
// ---------------------------------------------------------------------------
__global__ void fold_w(const float* __restrict__ W, const float* __restrict__ ar,
                       float* __restrict__ wf, int K, int H, int D) {
  int t = blockIdx.x * blockDim.x + threadIdx.x;
  if (t >= K * H) return;
  int k = t / H, h = t % H;
  const float* w = W + (long)k * H * D + h * D;
  const float* a = ar + h * D;
  float s = 0.f;
  for (int d = 0; d < D; ++d) s += w[d] * a[d];
  wf[t] = s;
}

// ---------------------------------------------------------------------------
// er[n,h] = sum_k x[n,k] * wf[k,h]            (dest-side logits, no fd matrix)
// ---------------------------------------------------------------------------
__global__ void er_dense(const float* __restrict__ x, const float* __restrict__ wf,
                         float* __restrict__ er, int N, int K, int H) {
  int t = blockIdx.x * blockDim.x + threadIdx.x;
  if (t >= N * H) return;
  int n = t / H, h = t % H;
  const float* xr = x + (long)n * K;
  float s = 0.f;
  for (int k = 0; k < K; ++k) s += xr[k] * wf[k * H + h];
  er[t] = s;
}

// ---------------------------------------------------------------------------
__global__ void fill_f(float* __restrict__ p, float v, long n) {
  long t = (long)blockIdx.x * blockDim.x + threadIdx.x;
  if (t < n) p[t] = v;
}

__global__ void fill_bias(float* __restrict__ out, const float* __restrict__ b,
                          long n, int M) {
  long t = (long)blockIdx.x * blockDim.x + threadIdx.x;
  if (t < n) out[t] = b[t % M];
}

// ---------------------------------------------------------------------------
// e = leaky_relu(el[src]+er[dst], 0.2); segment max over dst (atomic)
// ---------------------------------------------------------------------------
__global__ void edge_logits(const int* __restrict__ src, const int* __restrict__ dst,
                            const float* __restrict__ el, const float* __restrict__ er,
                            float* __restrict__ ebuf, float* __restrict__ mmax,
                            int E, int H) {
  int t = blockIdx.x * blockDim.x + threadIdx.x;
  if (t >= E * H) return;
  int e = t / H, h = t % H;
  float v = el[src[e] * H + h] + er[dst[e] * H + h];
  v = (v > 0.f) ? v : 0.2f * v;
  ebuf[t] = v;
  atomic_max_float(&mmax[dst[e] * H + h], v);
}

// ex = exp(e - m[dst]); den[dst] += ex
__global__ void edge_exp(const int* __restrict__ dst, float* __restrict__ ebuf,
                         const float* __restrict__ mmax, float* __restrict__ den,
                         int E, int H) {
  int t = blockIdx.x * blockDim.x + threadIdx.x;
  if (t >= E * H) return;
  int e = t / H, h = t % H;
  float ex = expf(ebuf[t] - mmax[dst[e] * H + h]);
  ebuf[t] = ex;
  atomicAdd(&den[dst[e] * H + h], ex);
}

// out[dst, c] += pfs[src, c] * (ex[e,h]/den[dst,h]),  h = c / D
__global__ void edge_aggregate(const int* __restrict__ src, const int* __restrict__ dst,
                               const float* __restrict__ pfs, const float* __restrict__ ebuf,
                               const float* __restrict__ den, float* __restrict__ out,
                               int E, int H, int D, int M) {
  long t = (long)blockIdx.x * blockDim.x + threadIdx.x;
  if (t >= (long)E * M) return;
  int e = (int)(t / M), c = (int)(t % M);
  int h = c / D;
  int d = dst[e];
  float alpha = ebuf[e * H + h] / den[d * H + h];
  atomicAdd(&out[(long)d * M + c], pfs[(long)src[e] * M + c] * alpha);
}

// ---------------------------------------------------------------------------
// BatchNorm batch statistics (biased variance) + activation
// ---------------------------------------------------------------------------
#define BN_ROWS 100
__global__ void bn_stats(const float* __restrict__ x, float* __restrict__ sum,
                         float* __restrict__ sumsq, int N, int M) {
  int c = threadIdx.x;
  if (c >= M) return;
  int r0 = blockIdx.x * BN_ROWS;
  float s = 0.f, s2 = 0.f;
  for (int r = 0; r < BN_ROWS && (r0 + r) < N; ++r) {
    float v = x[(long)(r0 + r) * M + c];
    s += v; s2 += v * v;
  }
  atomicAdd(&sum[c], s);
  atomicAdd(&sumsq[c], s2);
}

__global__ void bn_apply(float* __restrict__ x, const float* __restrict__ sum,
                         const float* __restrict__ sumsq, const float* __restrict__ g,
                         const float* __restrict__ b, int N, int M, int act) {
  long t = (long)blockIdx.x * blockDim.x + threadIdx.x;
  if (t >= (long)N * M) return;
  int c = (int)(t % M);
  float mu  = sum[c] / (float)N;
  float var = sumsq[c] / (float)N - mu * mu;
  float y = g[c] * (x[t] - mu) * rsqrtf(var + BN_EPS) + b[c];
  y = (act == 0) ? ((y > 0.f) ? y : 0.01f * y) : tanhf(y);
  x[t] = y;
}

// ---------------------------------------------------------------------------
// Host-side helpers
// ---------------------------------------------------------------------------
static void run_gat_rel(hipStream_t stream,
                        const float* x_src, int Ns, const float* x_dst, int Nd, int K,
                        const float* W, const float* al, const float* ar, const float* bias,
                        const int* src, const int* dst, int H, int D, int M,
                        float* pfs, float* el, float* er, float* wf,
                        float* mmax, float* den, float* ebuf, float* out) {
  // source projection via WMMA
  gemm_wmma_f16<<<dim3(Ns / 32, M / 64), 128, 0, stream>>>(x_src, W, pfs, Ns, K, M);
  // source-side logits reduced from projection
  el_reduce<<<(Ns * H + 255) / 256, 256, 0, stream>>>(pfs, al, el, Ns, H, D);
  // dest-side logits via folded weights (fd never materialized)
  fold_w<<<(K * H + 255) / 256, 256, 0, stream>>>(W, ar, wf, K, H, D);
  er_dense<<<(Nd * H + 255) / 256, 256, 0, stream>>>(x_dst, wf, er, Nd, K, H);
  // edge softmax
  fill_f<<<(Nd * H + 255) / 256, 256, 0, stream>>>(mmax, -INFINITY, (long)Nd * H);
  fill_f<<<(Nd * H + 255) / 256, 256, 0, stream>>>(den, 0.f, (long)Nd * H);
  edge_logits<<<(N_EDGE * H + 255) / 256, 256, 0, stream>>>(src, dst, el, er, ebuf, mmax, N_EDGE, H);
  edge_exp<<<(N_EDGE * H + 255) / 256, 256, 0, stream>>>(dst, ebuf, mmax, den, N_EDGE, H);
  // aggregate (out initialized to bias; isolated nodes end up = bias, like ref)
  long no = (long)Nd * M;
  fill_bias<<<(int)((no + 255) / 256), 256, 0, stream>>>(out, bias, no, M);
  long tot = (long)N_EDGE * M;
  edge_aggregate<<<(int)((tot + 255) / 256), 256, 0, stream>>>(src, dst, pfs, ebuf, den, out,
                                                              N_EDGE, H, D, M);
}

static void run_bn_act(hipStream_t stream, float* x, const float* g, const float* b,
                       int N, int M, int act, float* bsum, float* bsq) {
  fill_f<<<(M + 255) / 256, 256, 0, stream>>>(bsum, 0.f, (long)M);
  fill_f<<<(M + 255) / 256, 256, 0, stream>>>(bsq, 0.f, (long)M);
  bn_stats<<<(N + BN_ROWS - 1) / BN_ROWS, M, 0, stream>>>(x, bsum, bsq, N, M);
  long t = (long)N * M;
  bn_apply<<<(int)((t + 255) / 256), 256, 0, stream>>>(x, bsum, bsq, g, b, N, M, act);
}

extern "C" void kernel_launch(void* const* d_in, const int* in_sizes, int n_in,
                              void* d_out, int out_size, void* d_ws, size_t ws_size,
                              hipStream_t stream) {
  const float* user_feat = (const float*)d_in[0];
  const float* item_feat = (const float*)d_in[1];
  const float* Wl[4]  = {(const float*)d_in[2],  (const float*)d_in[6],
                         (const float*)d_in[10], (const float*)d_in[14]};
  const float* all_[4] = {(const float*)d_in[3],  (const float*)d_in[7],
                          (const float*)d_in[11], (const float*)d_in[15]};
  const float* arl[4] = {(const float*)d_in[4],  (const float*)d_in[8],
                         (const float*)d_in[12], (const float*)d_in[16]};
  const float* bl[4]  = {(const float*)d_in[5],  (const float*)d_in[9],
                         (const float*)d_in[13], (const float*)d_in[17]};
  const float* gamma_u = (const float*)d_in[18];
  const float* beta_u  = (const float*)d_in[19];
  const float* gamma_i = (const float*)d_in[20];
  const float* beta_i  = (const float*)d_in[21];
  const int* src_ui = (const int*)d_in[22];
  const int* dst_ui = (const int*)d_in[23];
  const int* src_iu = (const int*)d_in[24];
  const int* dst_iu = (const int*)d_in[25];

  // workspace layout (floats)
  float* ws = (float*)d_ws;
  const size_t NB = (size_t)N_USER * HID;           // 10.24M floats per node buffer
  float* hu_a = ws;
  float* hi_a = ws + NB;
  float* hu_b = ws + 2 * NB;
  float* hi_b = ws + 3 * NB;
  float* pfs  = ws + 4 * NB;
  float* ebuf = ws + 5 * NB;                        // N_EDGE * 8 max
  float* el   = ebuf + (size_t)N_EDGE * 8;
  float* er   = el   + (size_t)N_USER * 8;
  float* mmax = er   + (size_t)N_USER * 8;
  float* den  = mmax + (size_t)N_USER * 8;
  float* wf   = den  + (size_t)N_USER * 8;          // K*H <= 256*8
  float* bsum = wf   + (size_t)HID * 8;
  float* bsq  = bsum + HID;

  const int Hs[3] = {8, 4, 4};
  const int Ks[3] = {IN_FEATS, HID, HID};

  const float* cu = user_feat;
  const float* ci = item_feat;

  for (int L = 0; L < 3; ++L) {
    const int H = Hs[L], K = Ks[L], M = HID, D = M / H;
    float* nu = (L % 2 == 0) ? hu_a : hu_b;
    float* ni = (L % 2 == 0) ? hi_a : hi_b;
    const int KW = K * M;   // per-relation W stride
    // relation 0: users -> items (targets items)
    run_gat_rel(stream, cu, N_USER, ci, N_ITEM, K,
                Wl[L], all_[L], arl[L], bl[L],
                src_ui, dst_ui, H, D, M, pfs, el, er, wf, mmax, den, ebuf, ni);
    // relation 1: items -> users (targets users)
    run_gat_rel(stream, ci, N_ITEM, cu, N_USER, K,
                Wl[L] + KW, all_[L] + H * D, arl[L] + H * D, bl[L] + M,
                src_iu, dst_iu, H, D, M, pfs, el, er, wf, mmax, den, ebuf, nu);
    const int act = (L == 2) ? 1 : 0;
    run_bn_act(stream, nu, gamma_u + L * HID, beta_u + L * HID, N_USER, M, act, bsum, bsq);
    run_bn_act(stream, ni, gamma_i + L * HID, beta_i + L * HID, N_ITEM, M, act, bsum, bsq);
    cu = nu; ci = ni;
  }

  // final output layer: H=2, D=64, M=128, no BN / activation, straight to d_out
  {
    const int H = 2, K = HID, M = OUTF, D = M / H, KW = K * M;
    float* hu_o = (float*)d_out;                            // first in return order
    float* hi_o = (float*)d_out + (size_t)N_USER * OUTF;    // second
    run_gat_rel(stream, cu, N_USER, ci, N_ITEM, K,
                Wl[3], all_[3], arl[3], bl[3],
                src_ui, dst_ui, H, D, M, pfs, el, er, wf, mmax, den, ebuf, hi_o);
    run_gat_rel(stream, ci, N_ITEM, cu, N_USER, K,
                Wl[3] + KW, all_[3] + H * D, arl[3] + H * D, bl[3] + M,
                src_iu, dst_iu, H, D, M, pfs, el, er, wf, mmax, den, ebuf, hu_o);
  }
}
